// MambaBlock_47536698032514
// MI455X (gfx1250) — matrix operand order repaired and checked by reading, not compiled, not run on hardware
//
#include <hip/hip_runtime.h>
#include <hip/hip_bf16.h>

typedef __attribute__((ext_vector_type(16))) _Float16 v16h;
typedef __attribute__((ext_vector_type(8)))  float    v8f;
typedef int v4i_vec __attribute__((vector_size(16)));   // int4 as clang vector type

#define D_MODEL 256
#define D_STATE 16
#define D_INNER 512
#define D_HID   256
#define B_BATCH 2
#define SEQ_L   4096
#define NROWS   (B_BATCH * SEQ_L)   /* 8192 */
#define LN_EPS  1e-5f

#define A_TILE (128 * 32)
#define B_TILE (64 * 32)

// CDNA5 async global->LDS copy (ASYNCcnt path), guarded per-toolchain.
#if defined(__AMDGCN__) && defined(__has_builtin)
#  if __has_builtin(__builtin_amdgcn_global_load_async_to_lds_b128) && \
      __has_builtin(__builtin_amdgcn_s_wait_asynccnt)
#    define USE_ASYNC_LDS 1
#  endif
#endif
#ifndef USE_ASYNC_LDS
#  define USE_ASYNC_LDS 0
#endif

// Builtin signature: (int4 addrspace(1)*, int4 addrspace(3)*, imm offset, imm cpol)
#define GAS1(p) ((__attribute__((address_space(1))) v4i_vec*)(p))
#define LAS3(p) ((__attribute__((address_space(3))) v4i_vec*)(p))

// ---------------------------------------------------------------- helpers
__device__ __forceinline__ float gelu_exact(float x) {
    return 0.5f * x * (1.0f + erff(x * 0.70710678118654752440f));
}
__device__ __forceinline__ float softplus_f(float x) {
    if (x > 20.0f) return x;
    return log1pf(__expf(x));
}

// ---------------------------------------------------------------- weights fp32 -> f16
__global__ void convert_weights_kernel(const float* __restrict__ w_in,
                                       const float* __restrict__ w_dt,
                                       const float* __restrict__ w_out,
                                       _Float16* __restrict__ w_in_h,
                                       _Float16* __restrict__ w_dt_h,
                                       _Float16* __restrict__ w_out_h) {
    int idx = blockIdx.x * blockDim.x + threadIdx.x;
    const int n1 = D_INNER * D_MODEL;   // 131072
    const int n2 = D_HID * D_HID;       // 65536
    const int n3 = D_MODEL * D_INNER;   // 131072
    if (idx < n1)                w_in_h[idx]             = (_Float16)w_in[idx];
    else if (idx < n1 + n2)      w_dt_h[idx - n1]        = (_Float16)w_dt[idx - n1];
    else if (idx < n1 + n2 + n3) w_out_h[idx - n1 - n2]  = (_Float16)w_out[idx - n1 - n2];
}

// ---------------------------------------------------------------- LayerNorm: x(b,c,L) -> xn_f16 (b*L, c)
__global__ void layernorm_kernel(const float* __restrict__ x,
                                 const float* __restrict__ g,
                                 const float* __restrict__ beta,
                                 _Float16* __restrict__ xn) {
    int wid  = threadIdx.x >> 5;
    int lane = threadIdx.x & 31;
    int row  = blockIdx.x * (blockDim.x >> 5) + wid;   // row = b*L + l
    if (row >= NROWS) return;
    int b = row >> 12;          // / 4096
    int l = row & (SEQ_L - 1);
    const float* xb = x + (size_t)b * D_MODEL * SEQ_L + l;
    float v[8];
    float s = 0.f, sq = 0.f;
#pragma unroll
    for (int i = 0; i < 8; ++i) {
        int c = lane + 32 * i;
        v[i] = xb[(size_t)c * SEQ_L];
        s  += v[i];
        sq += v[i] * v[i];
    }
#pragma unroll
    for (int off = 16; off >= 1; off >>= 1) {
        s  += __shfl_xor(s,  off, 32);
        sq += __shfl_xor(sq, off, 32);
    }
    float mu   = s  * (1.0f / D_MODEL);
    float var  = sq * (1.0f / D_MODEL) - mu * mu;
    float rstd = rsqrtf(var + LN_EPS);
    _Float16* outp = xn + (size_t)row * D_MODEL;
#pragma unroll
    for (int i = 0; i < 8; ++i) {
        int c = lane + 32 * i;
        outp[c] = (_Float16)((v[i] - mu) * rstd * g[c] + beta[c]);
    }
}

// ---------------------------------------------------------------- 128x64 WMMA GEMM core, double-buffered async LDS
// A (M x K, row-major, leading dim lda) * W^T, W is (N x K, row-major).
// Block = 256 threads (8 waves as 4x2). Wave tile = 32x32 (4 accumulators).
// Ping-pong LDS: async fill of tile k+1 overlaps WMMA on tile k.
__device__ __forceinline__ void gemm_core(const _Float16* __restrict__ Ag, int lda,
                                          const _Float16* __restrict__ Wg, int K,
                                          int blockM, int blockN,
                                          _Float16* Ah, _Float16* Bh,   // Ah: 2*A_TILE, Bh: 2*B_TILE
                                          v8f c[2][2]) {
    const int t      = threadIdx.x;
    const int lane   = t & 31;
    const int wid    = t >> 5;
    const int wave_m = wid & 3;        // 0..3 -> 32-row slab
    const int wave_n = wid >> 2;       // 0..1 -> 32-col slab
    const int r      = lane & 15;
    const int hf     = lane >> 4;
    const int ia0 = t >> 2,         sa0 = t & 3;         // A chunk 0: rows 0..63
    const int ia1 = (t + 256) >> 2, sa1 = t & 3;         // A chunk 1: rows 64..127
    const int ib  = t >> 2,         sb  = t & 3;         // B: rows(n) 0..63
#pragma unroll
    for (int p = 0; p < 2; ++p)
#pragma unroll
        for (int q = 0; q < 8; ++q) { c[p][0][q] = 0.f; c[p][1][q] = 0.f; }

    const int nk = K / 32;

#if USE_ASYNC_LDS
    auto issue_fill = [&](int buf, int k0) {
        _Float16* Ad = Ah + buf * A_TILE;
        _Float16* Bd = Bh + buf * B_TILE;
        __builtin_amdgcn_global_load_async_to_lds_b128(
            GAS1(Ag + (size_t)(blockM + ia0) * lda + k0 + sa0 * 8),
            LAS3(Ad + (size_t)t * 8), 0, 0);
        __builtin_amdgcn_global_load_async_to_lds_b128(
            GAS1(Ag + (size_t)(blockM + ia1) * lda + k0 + sa1 * 8),
            LAS3(Ad + (size_t)(t + 256) * 8), 0, 0);
        __builtin_amdgcn_global_load_async_to_lds_b128(
            GAS1(Wg + (size_t)(blockN + ib) * K + k0 + sb * 8),
            LAS3(Bd + (size_t)t * 8), 0, 0);
    };
    issue_fill(0, 0);                       // preamble: fill buffer 0
    for (int kk = 0; kk < nk; ++kk) {
        const int cur = kk & 1;
        const _Float16* Ac = Ah + cur * A_TILE;
        const _Float16* Bc = Bh + cur * B_TILE;
        __builtin_amdgcn_s_wait_asynccnt(0);   // my fills of buf[cur] landed
        __syncthreads();                       // all waves' fills landed; prior reads done
        if (kk + 1 < nk) issue_fill(cur ^ 1, (kk + 1) * 32);   // overlap with compute

        union { v16h v; uint32_t u[8]; } af[2], bf[2];
#pragma unroll
        for (int sm = 0; sm < 2; ++sm) {
            const _Float16* arow = Ac + (wave_m * 32 + sm * 16 + r) * 32;
            const uint32_t* a1p = (const uint32_t*)(arow + 8 * hf);
            const uint32_t* a2p = (const uint32_t*)(arow + 16 + 8 * hf);
#pragma unroll
            for (int q = 0; q < 4; ++q) { af[sm].u[q] = a1p[q]; af[sm].u[4 + q] = a2p[q]; }
        }
#pragma unroll
        for (int sn = 0; sn < 2; ++sn) {
            const uint32_t* bp = (const uint32_t*)(Bc + (wave_n * 32 + sn * 16 + r) * 32 + 8 * hf);
#pragma unroll
            for (int q = 0; q < 4; ++q) { bf[sn].u[q] = bp[q]; bf[sn].u[4 + q] = bp[8 + q]; }
        }
        c[0][0] = __builtin_amdgcn_wmma_f32_16x16x32_f16(false, af[0].v, false, bf[0].v,
                                                         (short)0, c[0][0], false, false);
        c[0][1] = __builtin_amdgcn_wmma_f32_16x16x32_f16(false, af[0].v, false, bf[1].v,
                                                         (short)0, c[0][1], false, false);
        c[1][0] = __builtin_amdgcn_wmma_f32_16x16x32_f16(false, af[1].v, false, bf[0].v,
                                                         (short)0, c[1][0], false, false);
        c[1][1] = __builtin_amdgcn_wmma_f32_16x16x32_f16(false, af[1].v, false, bf[1].v,
                                                         (short)0, c[1][1], false, false);
    }
#else
    for (int kk = 0; kk < nk; ++kk) {
        const int k0 = kk * 32;
        uint4 a0 = *(const uint4*)(Ag + (size_t)(blockM + ia0) * lda + k0 + sa0 * 8);
        uint4 a1 = *(const uint4*)(Ag + (size_t)(blockM + ia1) * lda + k0 + sa1 * 8);
        uint4 bv = *(const uint4*)(Wg + (size_t)(blockN + ib) * K + k0 + sb * 8);
        __syncthreads();
        ((uint4*)Ah)[t]       = a0;
        ((uint4*)Ah)[t + 256] = a1;
        ((uint4*)Bh)[t]       = bv;
        __syncthreads();
        union { v16h v; uint32_t u[8]; } af[2], bf[2];
#pragma unroll
        for (int sm = 0; sm < 2; ++sm) {
            const _Float16* arow = Ah + (wave_m * 32 + sm * 16 + r) * 32;
            const uint32_t* a1p = (const uint32_t*)(arow + 8 * hf);
            const uint32_t* a2p = (const uint32_t*)(arow + 16 + 8 * hf);
#pragma unroll
            for (int q = 0; q < 4; ++q) { af[sm].u[q] = a1p[q]; af[sm].u[4 + q] = a2p[q]; }
        }
#pragma unroll
        for (int sn = 0; sn < 2; ++sn) {
            const uint32_t* bp = (const uint32_t*)(Bh + (wave_n * 32 + sn * 16 + r) * 32 + 8 * hf);
#pragma unroll
            for (int q = 0; q < 4; ++q) { bf[sn].u[q] = bp[q]; bf[sn].u[4 + q] = bp[8 + q]; }
        }
        c[0][0] = __builtin_amdgcn_wmma_f32_16x16x32_f16(false, af[0].v, false, bf[0].v,
                                                         (short)0, c[0][0], false, false);
        c[0][1] = __builtin_amdgcn_wmma_f32_16x16x32_f16(false, af[0].v, false, bf[1].v,
                                                         (short)0, c[0][1], false, false);
        c[1][0] = __builtin_amdgcn_wmma_f32_16x16x32_f16(false, af[1].v, false, bf[0].v,
                                                         (short)0, c[1][0], false, false);
        c[1][1] = __builtin_amdgcn_wmma_f32_16x16x32_f16(false, af[1].v, false, bf[1].v,
                                                         (short)0, c[1][1], false, false);
    }
#endif
}

// C/D tile coordinates for (sub_m, sub_n, v): caller adds blockM/blockN.
#define EPILOGUE_COORDS                                  \
    int lane = threadIdx.x & 31, wid = threadIdx.x >> 5; \
    int wave_m = wid & 3, wave_n = wid >> 2;             \
    int r = lane & 15, hf = lane >> 4;

// ---------------------------------------------------------------- GEMM1: xr = xn @ w_in^T + b_in  (fp32, 8192x512)
__global__ void gemm1_kernel(const _Float16* __restrict__ xn,
                             const _Float16* __restrict__ w_in_h,
                             const float* __restrict__ b_in,
                             float* __restrict__ xr) {
    __shared__ alignas(16) _Float16 Ah[2 * A_TILE];
    __shared__ alignas(16) _Float16 Bh[2 * B_TILE];
    int blockM = blockIdx.x * 128, blockN = blockIdx.y * 64;
    v8f c[2][2];
    gemm_core(xn, D_MODEL, w_in_h, D_MODEL, blockM, blockN, Ah, Bh, c);
    EPILOGUE_COORDS
#pragma unroll
    for (int sm = 0; sm < 2; ++sm)
#pragma unroll
        for (int sn = 0; sn < 2; ++sn)
#pragma unroll
            for (int v = 0; v < 8; ++v) {
                int row = blockM + wave_m * 32 + sm * 16 + v + 8 * hf;
                int col = blockN + wave_n * 32 + sn * 16 + r;
                xr[(size_t)row * D_INNER + col] = c[sm][sn][v] + b_in[col];
            }
}

// ---------------------------------------------------------------- depthwise conv (k=4, pad 1/2) + GELU
// writes u_t (b, Dh, L) fp32 and the xa half of GEMM3's A matrix (cols 256..511) f16
__global__ void conv_gelu_kernel(const float* __restrict__ xr,
                                 const float* __restrict__ conv_w,
                                 const float* __restrict__ conv_b,
                                 float* __restrict__ u_t,
                                 _Float16* __restrict__ A2) {
    int idx = blockIdx.x * blockDim.x + threadIdx.x;    // d fastest for coalescing
    int d = idx & (D_HID - 1);
    int l = (idx >> 8) & (SEQ_L - 1);
    int b = idx >> 20;
    const float* xp = xr + (size_t)b * SEQ_L * D_INNER + d;   // column d of xp
    float acc = conv_b[d];
#pragma unroll
    for (int k = 0; k < 4; ++k) {
        int ls = l - 1 + k;
        float xv = (ls >= 0 && ls < SEQ_L) ? xp[(size_t)ls * D_INNER] : 0.f;
        acc += conv_w[d * 4 + k] * xv;
    }
    float xa = gelu_exact(acc);
    u_t[((size_t)(b * D_HID + d)) * SEQ_L + l] = xa;
    A2[((size_t)(b * SEQ_L + l)) * D_INNER + D_HID + d] = (_Float16)xa;
}

// ---------------------------------------------------------------- GEMM2: dl = softplus(gelu(xa @ w_dt^T + b_dt)), stored transposed (b, Dh, L)
__global__ void gemm2_kernel(const _Float16* __restrict__ A2,
                             const _Float16* __restrict__ w_dt_h,
                             const float* __restrict__ b_dt,
                             float* __restrict__ dl_t) {
    __shared__ alignas(16) _Float16 Ah[2 * A_TILE];
    __shared__ alignas(16) _Float16 Bh[2 * B_TILE];
    int blockM = blockIdx.x * 128, blockN = blockIdx.y * 64;
    v8f c[2][2];
    gemm_core(A2 + D_HID, D_INNER, w_dt_h, D_HID, blockM, blockN, Ah, Bh, c);
    EPILOGUE_COORDS
#pragma unroll
    for (int sm = 0; sm < 2; ++sm)
#pragma unroll
        for (int sn = 0; sn < 2; ++sn)
#pragma unroll
            for (int v = 0; v < 8; ++v) {
                int row = blockM + wave_m * 32 + sm * 16 + v + 8 * hf;   // b*L + l
                int b = row >> 12;
                int l = row & (SEQ_L - 1);
                int col = blockN + wave_n * 32 + sn * 16 + r;
                float dv = softplus_f(gelu_exact(c[sm][sn][v] + b_dt[col]));
                dl_t[((size_t)(b * D_HID + col)) * SEQ_L + l] = dv;
            }
}

// ---------------------------------------------------------------- selective scan
// lane = (d pair, n): dloc = lane>>4, n = lane&15. One wave handles 2 channels.
// h = exp(dl*A)*h + (dl*u)*B ; y = sum_n(h*C) + u*D ; y -> f16 into A2 cols 0..255
__global__ void scan_kernel(const float* __restrict__ A,
                            const float* __restrict__ Bp,
                            const float* __restrict__ Cp,
                            const float* __restrict__ Dp,
                            const float* __restrict__ dl_t,
                            const float* __restrict__ u_t,
                            _Float16* __restrict__ A2) {
    int b    = blockIdx.x >> 4;
    int dblk = blockIdx.x & 15;
    int wid  = threadIdx.x >> 5;
    int lane = threadIdx.x & 31;
    int dloc = lane >> 4;
    int n    = lane & 15;
    int d    = dblk * 16 + wid * 2 + dloc;

    float a_dn = A[d * D_STATE + n];
    float b_dn = Bp[d * D_STATE + n];
    float c_dn = Cp[d * D_STATE + n];
    float dp_d = Dp[d];

    const float* dlp = dl_t + ((size_t)(b * D_HID + d)) * SEQ_L;
    const float* up  = u_t  + ((size_t)(b * D_HID + d)) * SEQ_L;
    _Float16* yout   = A2 + (size_t)b * SEQ_L * D_INNER + d;

    float h = 0.f;
    const int src = dloc << 4;                    // broadcast source lane of this d-group
    for (int l0 = 0; l0 < SEQ_L; l0 += 4) {
        float4 dl4 = make_float4(0.f, 0.f, 0.f, 0.f);
        float4 u4  = make_float4(0.f, 0.f, 0.f, 0.f);
        if (n == 0) {
            dl4 = *(const float4*)(dlp + l0);
            u4  = *(const float4*)(up  + l0);
            __builtin_prefetch(dlp + l0 + 32, 0, 0);
            __builtin_prefetch(up  + l0 + 32, 0, 0);
        }
#pragma unroll
        for (int j = 0; j < 4; ++j) {
            float dlj = (j == 0) ? dl4.x : (j == 1) ? dl4.y : (j == 2) ? dl4.z : dl4.w;
            float uj  = (j == 0) ? u4.x  : (j == 1) ? u4.y  : (j == 2) ? u4.z  : u4.w;
            float dlv = __shfl(dlj, src, 32);
            float uv  = __shfl(uj,  src, 32);
            h = __expf(dlv * a_dn) * h + (dlv * uv) * b_dn;
            float acc = h * c_dn;
#pragma unroll
            for (int off = 8; off >= 1; off >>= 1) acc += __shfl_xor(acc, off, 32);
            if (n == 0) {
                float y = acc + uv * dp_d;
                yout[((size_t)(l0 + j)) * D_INNER] = (_Float16)y;
            }
        }
    }
}

// ---------------------------------------------------------------- GEMM3: out = [y,xa] @ w_out^T + b_out + res, stored (b, c, L)
__global__ void gemm3_kernel(const _Float16* __restrict__ A2,
                             const _Float16* __restrict__ w_out_h,
                             const float* __restrict__ b_out,
                             const float* __restrict__ xr,
                             float* __restrict__ out) {
    __shared__ alignas(16) _Float16 Ah[2 * A_TILE];
    __shared__ alignas(16) _Float16 Bh[2 * B_TILE];
    int blockM = blockIdx.x * 128, blockN = blockIdx.y * 64;
    v8f c[2][2];
    gemm_core(A2, D_INNER, w_out_h, D_INNER, blockM, blockN, Ah, Bh, c);
    EPILOGUE_COORDS
#pragma unroll
    for (int sm = 0; sm < 2; ++sm)
#pragma unroll
        for (int sn = 0; sn < 2; ++sn)
#pragma unroll
            for (int v = 0; v < 8; ++v) {
                int row = blockM + wave_m * 32 + sm * 16 + v + 8 * hf;   // b*L + l
                int b = row >> 12;
                int l = row & (SEQ_L - 1);
                int col = blockN + wave_n * 32 + sn * 16 + r;
                float o = c[sm][sn][v] + b_out[col]
                        + xr[(size_t)row * D_INNER + D_HID + col];
                out[((size_t)(b * D_MODEL + col)) * SEQ_L + l] = o;
            }
}

// ---------------------------------------------------------------- launcher
extern "C" void kernel_launch(void* const* d_in, const int* in_sizes, int n_in,
                              void* d_out, int out_size, void* d_ws, size_t ws_size,
                              hipStream_t stream) {
    const float* x      = (const float*)d_in[0];
    const float* ln_g   = (const float*)d_in[1];
    const float* ln_b   = (const float*)d_in[2];
    const float* w_in   = (const float*)d_in[3];
    const float* b_in   = (const float*)d_in[4];
    const float* conv_w = (const float*)d_in[5];
    const float* conv_b = (const float*)d_in[6];
    const float* A      = (const float*)d_in[7];
    const float* Bp     = (const float*)d_in[8];
    const float* Cp     = (const float*)d_in[9];
    const float* Dp     = (const float*)d_in[10];
    const float* w_dt   = (const float*)d_in[11];
    const float* b_dt   = (const float*)d_in[12];
    const float* w_out  = (const float*)d_in[13];
    const float* b_out  = (const float*)d_in[14];
    float* out = (float*)d_out;

    char* p = (char*)d_ws;
    auto alloc = [&](size_t bytes) {
        char* r = p;
        p += (bytes + 255) & ~(size_t)255;
        return r;
    };
    _Float16* xn      = (_Float16*)alloc((size_t)NROWS * D_MODEL * 2);     // 4 MB
    _Float16* w_in_h  = (_Float16*)alloc((size_t)D_INNER * D_MODEL * 2);   // 256 KB
    _Float16* w_dt_h  = (_Float16*)alloc((size_t)D_HID * D_HID * 2);       // 128 KB
    _Float16* w_out_h = (_Float16*)alloc((size_t)D_MODEL * D_INNER * 2);   // 256 KB
    float*    xr      = (float*)   alloc((size_t)NROWS * D_INNER * 4);     // 16 MB
    _Float16* A2      = (_Float16*)alloc((size_t)NROWS * D_INNER * 2);     // 8 MB  ([y | xa] f16)
    float*    u_t     = (float*)   alloc((size_t)B_BATCH * D_HID * SEQ_L * 4); // 8 MB
    float*    dl_t    = (float*)   alloc((size_t)B_BATCH * D_HID * SEQ_L * 4); // 8 MB

    convert_weights_kernel<<<1280, 256, 0, stream>>>(w_in, w_dt, w_out,
                                                     w_in_h, w_dt_h, w_out_h);
    layernorm_kernel<<<NROWS / 8, 256, 0, stream>>>(x, ln_g, ln_b, xn);

    dim3 g1(NROWS / 128, D_INNER / 64);
    gemm1_kernel<<<g1, 256, 0, stream>>>(xn, w_in_h, b_in, xr);

    conv_gelu_kernel<<<(B_BATCH * SEQ_L * D_HID) / 256, 256, 0, stream>>>(
        xr, conv_w, conv_b, u_t, A2);

    dim3 g2(NROWS / 128, D_HID / 64);
    gemm2_kernel<<<g2, 256, 0, stream>>>(A2, w_dt_h, b_dt, dl_t);

    scan_kernel<<<B_BATCH * (D_HID / 16), 256, 0, stream>>>(
        A, Bp, Cp, Dp, dl_t, u_t, A2);

    dim3 g3(NROWS / 128, D_MODEL / 64);
    gemm3_kernel<<<g3, 256, 0, stream>>>(A2, w_out_h, b_out, xr, out);
}
